// LSTMDecoder_81415400063533
// MI455X (gfx1250) — matrix operand order repaired and checked by reading, not compiled, not run on hardware
//
#include <hip/hip_runtime.h>
#include <hip/hip_bf16.h>
#include <math.h>

#define DEV __device__ __forceinline__

// Problem dimensions (fixed by the reference)
constexpr int kB = 32;      // batch
constexpr int kT = 64;      // decode steps
constexpr int kS = 64;      // source length
constexpr int kH = 512;     // hidden
constexpr int kE = 512;     // embed
constexpr int kV = 32000;   // vocab

typedef __attribute__((ext_vector_type(2)))  float          v2f;
typedef __attribute__((ext_vector_type(8)))  float          v8f;
typedef __attribute__((ext_vector_type(16))) __bf16         v16bf;
typedef __attribute__((ext_vector_type(8)))  unsigned short v8us;
typedef __attribute__((ext_vector_type(16))) unsigned short v16us;
typedef __attribute__((ext_vector_type(4)))  unsigned int   u32x4;
typedef __attribute__((ext_vector_type(8)))  unsigned int   u32x8;

DEV unsigned short f32_to_bf16(float f) {
    unsigned int u = __builtin_bit_cast(unsigned int, f);
    u += 0x7FFFu + ((u >> 16) & 1u);   // round-to-nearest-even
    return (unsigned short)(u >> 16);
}

DEV v8f wmma_f32(v2f a, v2f b, v8f c) {
    // D = A(16x4 f32) * B(4x16 f32) + C, full fp32 matrix path
    return __builtin_amdgcn_wmma_f32_16x16x4_f32(
        false, a, false, b, (short)0, c, false, false);
}

DEV v8f wmma_bf16(v16bf a, v16bf b, v8f c) {
    return __builtin_amdgcn_wmma_f32_16x16x32_bf16(
        false, a, false, b, (short)0, c, false, false);
}

DEV float sigm(float x) { return 1.0f / (1.0f + expf(-x)); }

// ---------------------------------------------------------------------------
// Setup kernels
// ---------------------------------------------------------------------------
__global__ void init_state_kernel(const float* __restrict__ h0,
                                  const float* __restrict__ c0,
                                  float* __restrict__ h,
                                  float* __restrict__ c,
                                  float* __restrict__ feed0) {
    int idx = blockIdx.x * blockDim.x + threadIdx.x;
    if (idx < kB * kH) {
        h[idx] = h0[idx];
        c[idx] = c0[idx];
        feed0[idx] = 0.0f;
    }
}

__global__ void convert_w_kernel(const float* __restrict__ w,
                                 unsigned short* __restrict__ wb) {
    size_t stride = (size_t)gridDim.x * blockDim.x;
    for (size_t i = (size_t)blockIdx.x * blockDim.x + threadIdx.x;
         i < (size_t)kV * kH; i += stride)
        wb[i] = f32_to_bf16(w[i]);
}

__global__ void embed_kernel(const int* __restrict__ tok,       // [B,T]
                             const float* __restrict__ emb,     // [V,E]
                             float* __restrict__ x) {           // [T,B,E]
    size_t idx = (size_t)blockIdx.x * blockDim.x + threadIdx.x;
    if (idx >= (size_t)kT * kB * kE) return;
    int e  = (int)(idx % kE);
    int bb = (int)((idx / kE) % kB);
    int t  = (int)(idx / ((size_t)kE * kB));
    int token = tok[bb * kT + t];
    x[idx] = emb[(size_t)token * kE + e];
}

// ---------------------------------------------------------------------------
// Step kernel 1: gates = [x_t | feed] @ w_ih^T + h @ w_hh^T + b_ih + b_hh
// f32 WMMA, M=32 (2 tiles), N=2048 (128 tiles), K=512+512+512
// ---------------------------------------------------------------------------
__global__ void gates_kernel(const float* __restrict__ x,     // [T,B,E]
                             const float* __restrict__ feed,  // [B,H]
                             const float* __restrict__ h,     // [B,H]
                             const float* __restrict__ w_ih,  // [4H, E+H]
                             const float* __restrict__ w_hh,  // [4H, H]
                             const float* __restrict__ b_ih,
                             const float* __restrict__ b_hh,
                             float* __restrict__ gates,       // [B, 4H]
                             int t) {
    const int wave = (blockIdx.x * blockDim.x + threadIdx.x) >> 5;
    const int lane = threadIdx.x & 31;
    const int tile_n = (wave & 127) * 16;        // 128 n-tiles
    const int tile_m = (wave >> 7) * 16;         // 2 m-tiles
    const int half = lane >> 4;
    const int mrow = tile_m + (lane & 15);       // A row (ISA 32-bit A layout)
    const int ncol = tile_n + (lane & 15);       // B col (row-striped B layout)
    const float* xt = x + (size_t)t * kB * kE;
    const float* wi = w_ih + (size_t)ncol * (kE + kH);
    const float* wh = w_hh + (size_t)ncol * kH;

    v8f acc = {};
    for (int kk = 0; kk < kE; kk += 4) {         // x_t against w_ih[:, :E]
        int k0 = kk + 2 * half;
        v2f a, b;
        a.x = xt[mrow * kE + k0];  a.y = xt[mrow * kE + k0 + 1];
        b.x = wi[k0];              b.y = wi[k0 + 1];
        acc = wmma_f32(a, b, acc);
    }
    for (int kk = 0; kk < kH; kk += 4) {         // feed against w_ih[:, E:]
        int k0 = kk + 2 * half;
        v2f a, b;
        a.x = feed[mrow * kH + k0];  a.y = feed[mrow * kH + k0 + 1];
        b.x = wi[kE + k0];           b.y = wi[kE + k0 + 1];
        acc = wmma_f32(a, b, acc);
    }
    for (int kk = 0; kk < kH; kk += 4) {         // h against w_hh
        int k0 = kk + 2 * half;
        v2f a, b;
        a.x = h[mrow * kH + k0];  a.y = h[mrow * kH + k0 + 1];
        b.x = wh[k0];             b.y = wh[k0 + 1];
        acc = wmma_f32(a, b, acc);
    }
    float bias = b_ih[ncol] + b_hh[ncol];
#pragma unroll
    for (int r = 0; r < 8; ++r) {
        int row = tile_m + r + 8 * half;         // C/D layout
        gates[row * (4 * kH) + ncol] = acc[r] + bias;
    }
}

// ---------------------------------------------------------------------------
// Step kernel 2: LSTM pointwise update (PyTorch gate order i,f,g,o)
// ---------------------------------------------------------------------------
__global__ void lstm_pointwise_kernel(const float* __restrict__ gates,
                                      float* __restrict__ h,
                                      float* __restrict__ c) {
    int idx = blockIdx.x * blockDim.x + threadIdx.x;
    if (idx >= kB * kH) return;
    int b = idx / kH, hh = idx % kH;
    const float* g = gates + (size_t)b * 4 * kH;
    float ig = g[hh], fg = g[kH + hh], gg = g[2 * kH + hh], og = g[3 * kH + hh];
    float cn = sigm(fg) * c[idx] + sigm(ig) * tanhf(gg);
    float hn = sigm(og) * tanhf(cn);
    c[idx] = cn;
    h[idx] = hn;
}

// ---------------------------------------------------------------------------
// Step kernel 3: q = h @ attn_in_w^T   (M=32, N=512, K=512, f32 WMMA)
// ---------------------------------------------------------------------------
__global__ void q_gemm_kernel(const float* __restrict__ h,
                              const float* __restrict__ w,   // [H,H]
                              float* __restrict__ q) {
    const int wave = (blockIdx.x * blockDim.x + threadIdx.x) >> 5;
    const int lane = threadIdx.x & 31;
    const int tile_n = (wave & 31) * 16;
    const int tile_m = (wave >> 5) * 16;
    const int half = lane >> 4;
    const int mrow = tile_m + (lane & 15);
    const int ncol = tile_n + (lane & 15);
    const float* wr = w + (size_t)ncol * kH;
    v8f acc = {};
    for (int kk = 0; kk < kH; kk += 4) {
        int k0 = kk + 2 * half;
        v2f a, b;
        a.x = h[mrow * kH + k0];  a.y = h[mrow * kH + k0 + 1];
        b.x = wr[k0];             b.y = wr[k0 + 1];
        acc = wmma_f32(a, b, acc);
    }
#pragma unroll
    for (int r = 0; r < 8; ++r)
        q[(tile_m + r + 8 * half) * kH + ncol] = acc[r];
}

// ---------------------------------------------------------------------------
// Step kernel 4: scores -> softmax over S -> ctx, one block per batch row
// ---------------------------------------------------------------------------
__global__ void attn_kernel(const float* __restrict__ enc,          // [S,B,H]
                            const unsigned char* __restrict__ mask, // [S,B]
                            const float* __restrict__ q,            // [B,H]
                            float* __restrict__ ctx,                // [B,H]
                            float* __restrict__ attn_out,           // [B,T,S]
                            int t) {
    __shared__ float part[kS][4];
    __shared__ float sc[kS];
    __shared__ float aw[kS];
    __shared__ float mxsum[2];
    const int b = blockIdx.x;
    const int tid = threadIdx.x;   // 256 threads

    {   // scores[s] = enc[s,b,:] . q[b,:]  (4 threads per s, 128 elems each)
        int s = tid >> 2, p = tid & 3;
        const float* er = enc + ((size_t)s * kB + b) * kH;
        const float* qr = q + b * kH;
        float acc = 0.0f;
        for (int i = p * 128; i < p * 128 + 128; ++i) acc += er[i] * qr[i];
        part[s][p] = acc;
    }
    __syncthreads();
    if (tid < kS) {
        float v = part[tid][0] + part[tid][1] + part[tid][2] + part[tid][3];
        if (mask[tid * kB + b]) v = -INFINITY;
        sc[tid] = v;
    }
    __syncthreads();
    if (tid == 0) {
        float m = sc[0];
        for (int s = 1; s < kS; ++s) m = fmaxf(m, sc[s]);
        float sum = 0.0f;
        for (int s = 0; s < kS; ++s) sum += expf(sc[s] - m);
        mxsum[0] = m; mxsum[1] = sum;
    }
    __syncthreads();
    if (tid < kS) {
        float a = expf(sc[tid] - mxsum[0]) / mxsum[1];
        aw[tid] = a;
        attn_out[((size_t)b * kT + t) * kS + tid] = a;
    }
    __syncthreads();
    for (int hh = tid; hh < kH; hh += 256) {
        float acc = 0.0f;
        for (int s = 0; s < kS; ++s)
            acc += aw[s] * enc[((size_t)s * kB + b) * kH + hh];
        ctx[b * kH + hh] = acc;
    }
}

// ---------------------------------------------------------------------------
// Step kernel 5: out = tanh([ctx|h] @ attn_out_w^T)  (M=32, N=512, K=1024)
// Also stores the bf16 copy of y in [B,T,H] order for the vocab GEMM.
// ---------------------------------------------------------------------------
__global__ void out_gemm_kernel(const float* __restrict__ ctx,   // [B,H]
                                const float* __restrict__ h,     // [B,H]
                                const float* __restrict__ w,     // [H, 2H]
                                float* __restrict__ outs,        // [T,B,H]
                                unsigned short* __restrict__ yb, // [B*T, H] bf16
                                int t) {
    const int wave = (blockIdx.x * blockDim.x + threadIdx.x) >> 5;
    const int lane = threadIdx.x & 31;
    const int tile_n = (wave & 31) * 16;
    const int tile_m = (wave >> 5) * 16;
    const int half = lane >> 4;
    const int mrow = tile_m + (lane & 15);
    const int ncol = tile_n + (lane & 15);
    const float* wr = w + (size_t)ncol * (2 * kH);
    v8f acc = {};
    for (int kk = 0; kk < kH; kk += 4) {           // ctx half
        int k0 = kk + 2 * half;
        v2f a, b;
        a.x = ctx[mrow * kH + k0];  a.y = ctx[mrow * kH + k0 + 1];
        b.x = wr[k0];               b.y = wr[k0 + 1];
        acc = wmma_f32(a, b, acc);
    }
    for (int kk = 0; kk < kH; kk += 4) {           // h half
        int k0 = kk + 2 * half;
        v2f a, b;
        a.x = h[mrow * kH + k0];  a.y = h[mrow * kH + k0 + 1];
        b.x = wr[kH + k0];        b.y = wr[kH + k0 + 1];
        acc = wmma_f32(a, b, acc);
    }
#pragma unroll
    for (int r = 0; r < 8; ++r) {
        int row = tile_m + r + 8 * half;
        float v = tanhf(acc[r]);
        outs[((size_t)t * kB + row) * kH + ncol] = v;          // feeds t+1
        yb[((size_t)row * kT + t) * kH + ncol] = f32_to_bf16(v);
    }
}

// ---------------------------------------------------------------------------
// Final vocab projection: logits = y_bf16 @ w_bf16^T + bias
//   Block = 8 waves, tile M=32 x N=256. The 32x512 bf16 A-panel is staged
//   into LDS ONCE per block by the Tensor Data Mover (tensor_load_to_lds),
//   with D# padding (4 dwords per 256-dword row -> LDS row stride 260 dwords
//   = 65 x u32x4). The WMMA 16-bit A-fragment is two CONTIGUOUS 16-byte
//   chunks per lane, so each fragment is exactly two ds_load_b128. The LDS
//   buffer's address is passed INTO the asm as an operand so the optimizer
//   knows the TDM writes it (otherwise GlobalOpt folds the never-stored LDS
//   global's loads to undef and deletes the ds_loads). Each wave computes a
//   2x2 grid of 16x16 tiles: 4 WMMAs per K-step from 2 A-frags (LDS) +
//   2 B-frags (global, L2-resident weights).
// ---------------------------------------------------------------------------
constexpr int kRowV4 = 65;   // padded LDS row stride in u32x4 units (260 dw)

__global__ void logits_kernel(const unsigned short* __restrict__ yb, // [2048,H]
                              const unsigned short* __restrict__ wb, // [V,H]
                              const float* __restrict__ bias,        // [V]
                              float* __restrict__ out) {             // [2048,V]
    __shared__ u32x4 lA[32 * kRowV4];   // 33,280 B (sole LDS alloc -> offset 0)

    const int bm = blockIdx.x / (kV / 256);      // 64 M-blocks
    const int bn = blockIdx.x % (kV / 256);      // 125 N-blocks
    const int tile_m0 = bm * 32;
    const int wave = threadIdx.x >> 5;           // 0..7
    const int lane = threadIdx.x & 31;
    const int half = lane >> 4;
    const int nbase = bn * 256 + wave * 32;      // this wave's 32-col strip

    // --- TDM: stage A panel yb[tile_m0 .. tile_m0+31][0..511] into LDS -----
    if (wave == 0) {
        unsigned long long ga =
            (unsigned long long)(const void*)yb + (unsigned long long)tile_m0 * kH * 2ull;
        u32x4 g0;
        g0[0] = 1u;                                   // count = 1 valid descriptor
        g0[1] = 0u;                                   // lds_addr = 0 (lA)
        g0[2] = (unsigned)(ga & 0xFFFFFFFFull);       // global_addr[31:0]
        g0[3] = (unsigned)((ga >> 32) & 0x01FFFFFFull) | 0x80000000u; // addr[56:32], type=2
        u32x8 g1;
        // data_size=2B (code 1), pad_enable, pad_interval=256 dw (code 7),
        // pad_amount=4 dw (code 3), workgroup_mask=0 (not in cluster)
        g1[0] = (1u << 16) | (1u << 20) | (7u << 22) | (3u << 25);
        g1[1] = ((unsigned)kH & 0xFFFFu) << 16;        // tensor_dim0[15:0] = 512
        g1[2] = ((unsigned)(kB * kT) & 0xFFFFu) << 16; // tensor_dim1[15:0] = 2048
        g1[3] = ((unsigned)kH) << 16;                  // tile_dim0 = 512
        g1[4] = 32u;                                   // tile_dim1 = 32 rows
        g1[5] = (unsigned)kH;                          // tensor_dim0_stride = 512
        g1[6] = 0u;
        g1[7] = 0u;
        // %2 (lA's address) is not referenced in the template: it exists so
        // the LDS global escapes into the asm and its loads are not folded
        // to undef by GlobalOpt (the TDM genuinely writes this buffer).
        asm volatile("tensor_load_to_lds %0, %1"
                     :: "s"(g0), "s"(g1),
                        "s"((unsigned long long)(size_t)&lA[0])
                     : "memory");
        __builtin_amdgcn_s_wait_tensorcnt(0);
    }
    __syncthreads();

    // --- 2x2 register tiles per wave ---------------------------------------
    const int arow0 = (lane & 15) * kRowV4;           // m-tile 0 row (vec4 units)
    const int arow1 = arow0 + 16 * kRowV4;            // m-tile 1 row
    const unsigned short* b0p = wb + (size_t)(nbase + (lane & 15)) * kH;
    const unsigned short* b1p = wb + (size_t)(nbase + 16 + (lane & 15)) * kH;

    v8f c00 = {}, c01 = {}, c10 = {}, c11 = {};
    for (int kk = 0; kk < kH; kk += 32) {
        // A-fragment = two contiguous 16B chunks per lane:
        //   chunk0: K = kk + 16*half     .. +7   (elements 0..7)
        //   chunk1: K = kk + 16*half + 8 .. +7   (elements 8..15)
        const int v0 = (kk >> 3) + 2 * half;          // vec4 offset within row
        v8us a0lo = __builtin_bit_cast(v8us, lA[arow0 + v0]);      // ds_load_b128
        v8us a0hi = __builtin_bit_cast(v8us, lA[arow0 + v0 + 1]);
        v8us a1lo = __builtin_bit_cast(v8us, lA[arow1 + v0]);
        v8us a1hi = __builtin_bit_cast(v8us, lA[arow1 + v0 + 1]);
        v16us a0u = __builtin_shufflevector(a0lo, a0hi,
            0, 1, 2, 3, 4, 5, 6, 7, 8, 9, 10, 11, 12, 13, 14, 15);
        v16us a1u = __builtin_shufflevector(a1lo, a1hi,
            0, 1, 2, 3, 4, 5, 6, 7, 8, 9, 10, 11, 12, 13, 14, 15);
        // B layout: lane holds one column, 16 contiguous K at kk + 16*half
        v16us b0u = *reinterpret_cast<const v16us*>(b0p + kk + 16 * half);
        v16us b1u = *reinterpret_cast<const v16us*>(b1p + kk + 16 * half);
        v16bf a0 = __builtin_bit_cast(v16bf, a0u);
        v16bf a1 = __builtin_bit_cast(v16bf, a1u);
        v16bf b0 = __builtin_bit_cast(v16bf, b0u);
        v16bf b1 = __builtin_bit_cast(v16bf, b1u);
        c00 = wmma_bf16(a0, b0, c00);
        c01 = wmma_bf16(a0, b1, c01);
        c10 = wmma_bf16(a1, b0, c10);
        c11 = wmma_bf16(a1, b1, c11);
    }

    const int ncol0 = nbase + (lane & 15);
    const int ncol1 = ncol0 + 16;
    const float bs0 = bias[ncol0];
    const float bs1 = bias[ncol1];
#pragma unroll
    for (int r = 0; r < 8; ++r) {
        int row0 = tile_m0 + r + 8 * half;
        int row1 = row0 + 16;
        out[(size_t)row0 * kV + ncol0] = c00[r] + bs0;
        out[(size_t)row0 * kV + ncol1] = c01[r] + bs1;
        out[(size_t)row1 * kV + ncol0] = c10[r] + bs0;
        out[(size_t)row1 * kV + ncol1] = c11[r] + bs1;
    }
}

// ---------------------------------------------------------------------------
// Host launcher
// ---------------------------------------------------------------------------
extern "C" void kernel_launch(void* const* d_in, const int* in_sizes, int n_in,
                              void* d_out, int out_size, void* d_ws, size_t ws_size,
                              hipStream_t stream) {
    const int*   tokens   = (const int*)  d_in[0];   // [B,T]
    const float* enc      = (const float*)d_in[1];   // [S,B,H]
    const float* h0       = (const float*)d_in[2];   // [1,B,H]
    const float* c0       = (const float*)d_in[3];   // [1,B,H]
    const unsigned char* mask = (const unsigned char*)d_in[4]; // [S,B] bool
    const float* emb      = (const float*)d_in[5];   // [V,E]
    const float* w_ih     = (const float*)d_in[6];   // [4H, E+H]
    const float* w_hh     = (const float*)d_in[7];   // [4H, H]
    const float* b_ih     = (const float*)d_in[8];
    const float* b_hh     = (const float*)d_in[9];
    const float* attn_in_w  = (const float*)d_in[10]; // [H,H]
    const float* attn_out_w = (const float*)d_in[11]; // [H,2H]
    const float* fc_out_w   = (const float*)d_in[12]; // [V,H]
    const float* fc_out_b   = (const float*)d_in[13]; // [V]

    float* logits = (float*)d_out;                         // [B,T,V]
    float* attn   = logits + (size_t)kB * kT * kV;         // [B,T,S]

    // Workspace carve-up (all 256B aligned)
    char* ws = (char*)d_ws;
    size_t off = 0;
    auto carve = [&](size_t bytes) {
        char* p = ws + off;
        off += (bytes + 255) & ~(size_t)255;
        return p;
    };
    float* x      = (float*)carve((size_t)kT * kB * kE * 4);  // [T,B,E]
    float* outs   = (float*)carve((size_t)kT * kB * kH * 4);  // [T,B,H]
    float* hbuf   = (float*)carve((size_t)kB * kH * 4);
    float* cbuf   = (float*)carve((size_t)kB * kH * 4);
    float* feed0  = (float*)carve((size_t)kB * kH * 4);
    float* gates  = (float*)carve((size_t)kB * 4 * kH * 4);
    float* qbuf   = (float*)carve((size_t)kB * kH * 4);
    float* ctxbuf = (float*)carve((size_t)kB * kH * 4);
    unsigned short* ybf = (unsigned short*)carve((size_t)kB * kT * kH * 2);
    unsigned short* wbf = (unsigned short*)carve((size_t)kV * kH * 2);
    (void)ws_size; (void)in_sizes; (void)n_in; (void)out_size;

    // Setup
    init_state_kernel<<<(kB * kH + 255) / 256, 256, 0, stream>>>(
        h0, c0, hbuf, cbuf, feed0);
    convert_w_kernel<<<4096, 256, 0, stream>>>(fc_out_w, wbf);
    embed_kernel<<<((size_t)kT * kB * kE + 255) / 256, 256, 0, stream>>>(
        tokens, emb, x);

    // Sequential decode: 64 steps
    for (int t = 0; t < kT; ++t) {
        const float* feed = (t == 0) ? feed0 : outs + (size_t)(t - 1) * kB * kH;
        // 256 wave-tiles (128 N x 2 M), 4 waves/block
        gates_kernel<<<64, 128, 0, stream>>>(
            x, feed, hbuf, w_ih, w_hh, b_ih, b_hh, gates, t);
        lstm_pointwise_kernel<<<(kB * kH + 255) / 256, 256, 0, stream>>>(
            gates, hbuf, cbuf);
        // 64 wave-tiles (32 N x 2 M)
        q_gemm_kernel<<<16, 128, 0, stream>>>(hbuf, attn_in_w, qbuf);
        attn_kernel<<<kB, 256, 0, stream>>>(enc, mask, qbuf, ctxbuf, attn, t);
        out_gemm_kernel<<<16, 128, 0, stream>>>(
            ctxbuf, hbuf, attn_out_w, outs, ybf, t);
    }

    // Vocab projection: 64 M-blocks x 125 N-blocks, 256 threads (8 waves)
    logits_kernel<<<64 * (kV / 256), 256, 0, stream>>>(
        ybf, wbf, fc_out_b, logits);
}
